// SI_30777735643264
// MI455X (gfx1250) — matrix operand order repaired and checked by reading, not compiled, not run on hardware
//
#include <hip/hip_runtime.h>
#include <math.h>

// Problem constants (from reference): B=8, N=89, C=1024, H=512
#define B_   8
#define NN   89
#define CC   1024
#define HH   512
#define JT2  3          // ceil(89/32) 32-row tiles for the adaptor
#define RTOT (B_ * NN)  // 712 rows for the per-node MLPs

typedef __attribute__((ext_vector_type(8)))  float  v8f;
typedef __attribute__((ext_vector_type(16))) __bf16 v16bf;
typedef __attribute__((ext_vector_type(8)))  __bf16 v8bf;

static __device__ __forceinline__ v16bf bf_combine(v8bf lo, v8bf hi) {
  v16bf r;
#pragma unroll
  for (int e = 0; e < 8; ++e) { r[e] = lo[e]; r[e + 8] = hi[e]; }
  return r;
}

// A-fragment (16x32 bf16) gather from a row-major LDS tile [16][ld].
// Layout: lanes 0-15 row M=lane hold K {0..15}; lanes 16-31 hold K {16..31}.
// -> per lane: 16 contiguous bf16 at kbase + 16*(lane>=16).
static __device__ __forceinline__ v16bf afrag(const __bf16* base, int ld, int lane, int kbase) {
  const __bf16* p = base + (lane & 15) * ld + kbase + ((lane >> 4) << 4);
  v8bf lo = *(const v8bf*)(p);
  v8bf hi = *(const v8bf*)(p + 8);
  return bf_combine(lo, hi);
}

// B-fragment (32x16 bf16) from the pre-swizzled weight buffer: 512 elems per fragment,
// lane-l owns 16 contiguous bf16 (col = l%16, K-half = l/16) -> 2x global_load_b128.
static __device__ __forceinline__ v16bf bfrag(const __bf16* frag, int lane) {
  const __bf16* p = frag + lane * 16;
  v8bf lo = *(const v8bf*)(p);
  v8bf hi = *(const v8bf*)(p + 8);
  return bf_combine(lo, hi);
}

#define WMMA_BF16(A, Bm, Cm) \
  __builtin_amdgcn_wmma_f32_16x16x32_bf16(false, (A), false, (Bm), (short)0, (Cm), false, false)

// ---------------------------------------------------------------------------
// Pack a row-major f32 weight [K][Nout] into bf16 B-fragment layout:
// dst[((ntile*(K/32)+kstep)*32 + lane)*16 + e] = W[kstep*32 + 16*(lane>=16) + e][ntile*16 + lane%16]
// ---------------------------------------------------------------------------
__global__ __launch_bounds__(256) void pack_w_kernel(const float* __restrict__ W,
                                                     __bf16* __restrict__ dst,
                                                     int K, int Nout) {
  const int nk    = K >> 5;
  const int nfrag = (Nout >> 4) * nk;
  const int gid   = blockIdx.x * 256 + threadIdx.x;
  const int frag  = gid >> 5, lane = gid & 31;
  if (frag >= nfrag) return;
  const int ntile = frag / nk, ks = frag - ntile * nk;
  const int n  = (ntile << 4) + (lane & 15);
  const int kb = (ks << 5) + ((lane >> 4) << 4);
  __bf16* out = dst + ((size_t)frag * 32 + lane) * 16;
#pragma unroll
  for (int e = 0; e < 16; ++e)
    out[e] = (__bf16)W[(size_t)(kb + e) * Nout + n];
}

// ---------------------------------------------------------------------------
// Adaptor: per block (b, i, 32-row j-tile): s[b,j,i] = sigmoid(MLP(x_j * x_i)).
// 32 rows per block => each B fragment is loaded once and feeds TWO WMMAs
// (row-halves), halving L2 weight traffic vs a 16-row tile.
// GEMM1 32x1024x1024, GEMM2 32x512x1024 (bf16 WMMA); GEMM3 512->1 on waves 0-1.
// h1 is written back into the pA LDS buffer (all reads of pA complete first).
// ---------------------------------------------------------------------------
__global__ __launch_bounds__(256) void adaptor_kernel(
    const float*  __restrict__ x,
    const __bf16* __restrict__ Wa1p, const float* __restrict__ ba1,
    const __bf16* __restrict__ Wa2p, const float* __restrict__ ba2,
    const float*  __restrict__ Wa3,  const float* __restrict__ ba3,
    float* __restrict__ s_out) {
  __shared__ __align__(16) float  xi[CC];
  __shared__ __align__(16) __bf16 pA[32 * CC];   // p tile, then reused as h1
  __shared__ __align__(16) __bf16 h2[32 * HH];

  const int bid  = blockIdx.x;
  const int jt   = bid % JT2;
  const int i    = (bid / JT2) % NN;
  const int b    = bid / (JT2 * NN);
  const int tid  = threadIdx.x, lane = tid & 31, wave = tid >> 5;

  for (int c = tid; c < CC; c += 256) xi[c] = x[((size_t)b * NN + i) * CC + c];
  __syncthreads();
  for (int idx = tid; idx < 32 * CC; idx += 256) {
    const int r = idx >> 10, c = idx & (CC - 1);
    int j = jt * 32 + r; if (j >= NN) j = NN - 1;            // pad tile with dup row
    pA[idx] = (__bf16)(x[((size_t)b * NN + j) * CC + c] * xi[c]);
  }
  __syncthreads();

  v8f zero = {};
  const int rb = (lane >> 4) << 3;
  // ---- GEMM1: p[32x1024] @ Wa1 -> h1 (ReLU); 8 waves x 8 ntiles x 2 row-halves
  {
    v8f acc0[8], acc1[8];
#pragma unroll
    for (int t = 0; t < 8; ++t) { acc0[t] = zero; acc1[t] = zero; }
    for (int ks = 0; ks < 32; ++ks) {
      v16bf a0 = afrag(pA,           CC, lane, ks << 5);
      v16bf a1 = afrag(pA + 16 * CC, CC, lane, ks << 5);
#pragma unroll
      for (int t = 0; t < 8; ++t) {
        v16bf bm = bfrag(Wa1p + ((size_t)((wave * 8 + t) * 32 + ks)) * 512, lane);
        acc0[t] = WMMA_BF16(a0, bm, acc0[t]);   // B fragment reused by both
        acc1[t] = WMMA_BF16(a1, bm, acc1[t]);   // row-halves (register-resident)
      }
    }
    __syncthreads();                            // everyone done reading pA
#pragma unroll
    for (int t = 0; t < 8; ++t) {
      const int col = (wave * 8 + t) * 16 + (lane & 15);
      const float bias = ba1[col];
#pragma unroll
      for (int r = 0; r < 8; ++r) {
        float v0 = acc0[t][r] + bias;
        float v1 = acc1[t][r] + bias;
        pA[(rb + r) * CC + col]        = (__bf16)(v0 > 0.f ? v0 : 0.f);  // h1 rows 0-15
        pA[(rb + r + 16) * CC + col]   = (__bf16)(v1 > 0.f ? v1 : 0.f);  // h1 rows 16-31
      }
    }
  }
  __syncthreads();
  // ---- GEMM2: h1[32x1024] @ Wa2 -> h2 (ReLU); 8 waves x 4 ntiles x 2 row-halves
  {
    v8f acc0[4], acc1[4];
#pragma unroll
    for (int t = 0; t < 4; ++t) { acc0[t] = zero; acc1[t] = zero; }
    for (int ks = 0; ks < 32; ++ks) {
      v16bf a0 = afrag(pA,           CC, lane, ks << 5);
      v16bf a1 = afrag(pA + 16 * CC, CC, lane, ks << 5);
#pragma unroll
      for (int t = 0; t < 4; ++t) {
        v16bf bm = bfrag(Wa2p + ((size_t)((wave * 4 + t) * 32 + ks)) * 512, lane);
        acc0[t] = WMMA_BF16(a0, bm, acc0[t]);
        acc1[t] = WMMA_BF16(a1, bm, acc1[t]);
      }
    }
#pragma unroll
    for (int t = 0; t < 4; ++t) {
      const int col = (wave * 4 + t) * 16 + (lane & 15);
      const float bias = ba2[col];
#pragma unroll
      for (int r = 0; r < 8; ++r) {
        float v0 = acc0[t][r] + bias;
        float v1 = acc1[t][r] + bias;
        h2[(rb + r) * HH + col]      = (__bf16)(v0 > 0.f ? v0 : 0.f);
        h2[(rb + r + 16) * HH + col] = (__bf16)(v1 > 0.f ? v1 : 0.f);
      }
    }
  }
  __syncthreads();
  // ---- GEMM3 (512 -> 1) + sigmoid; waves 0-1 cover the 32 rows
  if (wave < 2) {
    const int row = (wave << 4) + (lane & 15), half = lane >> 4;
    float sum = 0.f;
    for (int k = 0; k < 256; ++k) {
      const int kk = half * 256 + k;
      sum = fmaf((float)h2[row * HH + kk], Wa3[kk], sum);
    }
    sum += __shfl_down(sum, 16);
    const int j = jt * 32 + row;
    if (half == 0 && j < NN) {
      const float sv = sum + ba3[0];
      s_out[((size_t)b * NN + j) * NN + i] = 1.f / (1.f + __expf(-sv));
    }
  }
}

// ---------------------------------------------------------------------------
// Aggregations (K=89, trivial flops): a_t[b,i,:] = sum_j s[b,j,i]*adjA[j,i]*x[b,j,:]
//                                     m_t[b,i,:] = x[b,i,:] * sum_j adjM[j,i]*x[b,j,:]
// ---------------------------------------------------------------------------
__global__ __launch_bounds__(256) void agg_kernel(
    const float* __restrict__ x, const float* __restrict__ s,
    const float* __restrict__ adjA, const float* __restrict__ adjM,
    float* __restrict__ a_t, float* __restrict__ m_t) {
  __shared__ float cA[NN], cM[NN];
  const int b = blockIdx.x / NN, i = blockIdx.x % NN;
  const int tid = threadIdx.x;
  if (tid < NN) {
    cA[tid] = s[((size_t)b * NN + tid) * NN + i] * adjA[tid * NN + i];
    cM[tid] = adjM[tid * NN + i];
  }
  __syncthreads();
  for (int c = tid; c < CC; c += 256) {
    float sa = 0.f, sm = 0.f;
    for (int j = 0; j < NN; ++j) {
      const float xv = x[((size_t)b * NN + j) * CC + c];
      sa = fmaf(cA[j], xv, sa);
      sm = fmaf(cM[j], xv, sm);
    }
    const size_t o = ((size_t)b * NN + i) * CC + c;
    a_t[o] = sa;
    m_t[o] = x[o] * sm;
  }
}

// ---------------------------------------------------------------------------
// Per-node MLPs: out = (x + mlp2(a_t;Wadd) + mlp2(m_t;Wmod)) / 3.
// Both branches accumulate into the same WMMA C registers.
// ---------------------------------------------------------------------------
__global__ __launch_bounds__(256) void mlp2_kernel(
    const float* __restrict__ x,
    const float* __restrict__ a_t, const float* __restrict__ m_t,
    const __bf16* __restrict__ Wadd1p, const float* __restrict__ badd1,
    const __bf16* __restrict__ Wadd2p, const float* __restrict__ badd2,
    const __bf16* __restrict__ Wmod1p, const float* __restrict__ bmod1,
    const __bf16* __restrict__ Wmod2p, const float* __restrict__ bmod2,
    float* __restrict__ out) {
  __shared__ __align__(16) __bf16 Ain[16 * CC];
  __shared__ __align__(16) __bf16 hB[16 * CC];
  const int tid = threadIdx.x, lane = tid & 31, wave = tid >> 5;
  const int r0 = blockIdx.x * 16;
  v8f zero = {};
  v8f acc[8];
#pragma unroll
  for (int t = 0; t < 8; ++t) acc[t] = zero;

  for (int br = 0; br < 2; ++br) {
    const float*  src = br ? m_t    : a_t;
    const __bf16* W1  = br ? Wmod1p : Wadd1p;
    const float*  b1  = br ? bmod1  : badd1;
    const __bf16* W2  = br ? Wmod2p : Wadd2p;

    for (int idx = tid; idx < 16 * CC; idx += 256) {
      const int r = idx >> 10, c = idx & (CC - 1);
      int rg = r0 + r; if (rg >= RTOT) rg = RTOT - 1;
      Ain[idx] = (__bf16)src[(size_t)rg * CC + c];
    }
    __syncthreads();

    v8f hacc[8];
#pragma unroll
    for (int t = 0; t < 8; ++t) hacc[t] = zero;
    for (int ks = 0; ks < 32; ++ks) {
      v16bf a = afrag(Ain, CC, lane, ks << 5);
#pragma unroll
      for (int t = 0; t < 8; ++t) {
        v16bf bm = bfrag(W1 + ((size_t)((wave * 8 + t) * 32 + ks)) * 512, lane);
        hacc[t] = WMMA_BF16(a, bm, hacc[t]);
      }
    }
    const int rb = (lane >> 4) << 3;
#pragma unroll
    for (int t = 0; t < 8; ++t) {
      const int col = (wave * 8 + t) * 16 + (lane & 15);
      const float bias = b1[col];
#pragma unroll
      for (int r = 0; r < 8; ++r) {
        float v = hacc[t][r] + bias;
        hB[(rb + r) * CC + col] = (__bf16)(v > 0.f ? v : 0.f);
      }
    }
    __syncthreads();
    for (int ks = 0; ks < 32; ++ks) {
      v16bf a = afrag(hB, CC, lane, ks << 5);
#pragma unroll
      for (int t = 0; t < 8; ++t) {
        v16bf bm = bfrag(W2 + ((size_t)((wave * 8 + t) * 32 + ks)) * 512, lane);
        acc[t] = WMMA_BF16(a, bm, acc[t]);
      }
    }
    __syncthreads();
  }

  const int rb = (lane >> 4) << 3;
#pragma unroll
  for (int t = 0; t < 8; ++t) {
    const int col = (wave * 8 + t) * 16 + (lane & 15);
    const float bsum = badd2[col] + bmod2[col];
#pragma unroll
    for (int r = 0; r < 8; ++r) {
      const int rg = r0 + rb + r;
      if (rg < RTOT) {
        const size_t o = (size_t)rg * CC + col;
        out[o] = (x[o] + acc[t][r] + bsum) * (1.f / 3.f);
      }
    }
  }
}

// ---------------------------------------------------------------------------
extern "C" void kernel_launch(void* const* d_in, const int* in_sizes, int n_in,
                              void* d_out, int out_size, void* d_ws, size_t ws_size,
                              hipStream_t stream) {
  (void)in_sizes; (void)n_in; (void)out_size; (void)ws_size;
  const float* x     = (const float*)d_in[0];
  const float* adjA  = (const float*)d_in[1];
  const float* adjM  = (const float*)d_in[2];
  const float* Wa1   = (const float*)d_in[3];
  const float* ba1   = (const float*)d_in[4];
  const float* Wa2   = (const float*)d_in[5];
  const float* ba2   = (const float*)d_in[6];
  const float* Wa3   = (const float*)d_in[7];
  const float* ba3   = (const float*)d_in[8];
  const float* Wadd1 = (const float*)d_in[9];
  const float* badd1 = (const float*)d_in[10];
  const float* Wadd2 = (const float*)d_in[11];
  const float* badd2 = (const float*)d_in[12];
  const float* Wmod1 = (const float*)d_in[13];
  const float* bmod1 = (const float*)d_in[14];
  const float* Wmod2 = (const float*)d_in[15];
  const float* bmod2 = (const float*)d_in[16];
  float* out = (float*)d_out;

  // Workspace carve-out (~17.6 MB total), 256B-aligned slices.
  char* w = (char*)d_ws;
  auto take = [&](size_t bytes) { char* p = w; w += (bytes + 255) & ~(size_t)255; return p; };
  float*  s_ws   = (float*)take((size_t)B_ * NN * NN * 4);
  float*  a_t    = (float*)take((size_t)B_ * NN * CC * 4);
  float*  m_t    = (float*)take((size_t)B_ * NN * CC * 4);
  __bf16* Wa1p   = (__bf16*)take((size_t)CC * CC * 2);
  __bf16* Wa2p   = (__bf16*)take((size_t)CC * HH * 2);
  __bf16* Wadd1p = (__bf16*)take((size_t)CC * CC * 2);
  __bf16* Wadd2p = (__bf16*)take((size_t)CC * CC * 2);
  __bf16* Wmod1p = (__bf16*)take((size_t)CC * CC * 2);
  __bf16* Wmod2p = (__bf16*)take((size_t)CC * CC * 2);

  auto packGrid = [](int K, int Nout) { return ((Nout / 16) * (K / 32) * 32 + 255) / 256; };
  pack_w_kernel<<<packGrid(CC, CC), 256, 0, stream>>>(Wa1,   Wa1p,   CC, CC);
  pack_w_kernel<<<packGrid(CC, HH), 256, 0, stream>>>(Wa2,   Wa2p,   CC, HH);
  pack_w_kernel<<<packGrid(CC, CC), 256, 0, stream>>>(Wadd1, Wadd1p, CC, CC);
  pack_w_kernel<<<packGrid(CC, CC), 256, 0, stream>>>(Wadd2, Wadd2p, CC, CC);
  pack_w_kernel<<<packGrid(CC, CC), 256, 0, stream>>>(Wmod1, Wmod1p, CC, CC);
  pack_w_kernel<<<packGrid(CC, CC), 256, 0, stream>>>(Wmod2, Wmod2p, CC, CC);

  adaptor_kernel<<<B_ * NN * JT2, 256, 0, stream>>>(x, Wa1p, ba1, Wa2p, ba2, Wa3, ba3, s_ws);
  agg_kernel<<<B_ * NN, 256, 0, stream>>>(x, s_ws, adjA, adjM, a_t, m_t);
  mlp2_kernel<<<(RTOT + 15) / 16, 256, 0, stream>>>(x, a_t, m_t,
                                                    Wadd1p, badd1, Wadd2p, badd2,
                                                    Wmod1p, bmod1, Wmod2p, bmod2, out);
}
